// BiARMA_53996328845506
// MI455X (gfx1250) — matrix operand order repaired and checked by reading, not compiled, not run on hardware
//
#include <hip/hip_runtime.h>
#include <hip/hip_bf16.h>

#define NN 50000
#define EE 800000
#define KSTACK 2

typedef __attribute__((ext_vector_type(16))) __bf16 v16bf;
typedef __attribute__((ext_vector_type(8)))  float  v8f;

// C[row, coff+col] = A[row, 0:FIN] @ W[0:FIN, col] (+ bias[col])
// A row-stride lda, W row-stride ldw (== fout of W), C row-stride ldc.
// grid.x = nrows/16, block = (32, fout/16); one wave per 16x16 output tile.
// FIN is compile-time so the K-loop fully unrolls into consecutive v_wmma ops.
template <int FIN>
__global__ void k_gemm_wmma(const float* __restrict__ A, int lda,
                            const float* __restrict__ W, int ldw,
                            const float* __restrict__ bias,
                            float* __restrict__ C, int ldc, int coff) {
  const int lane    = threadIdx.x;       // 0..31, wave32
  const int colTile = threadIdx.y;
  const int row0    = blockIdx.x * 16;
  const int half    = lane >> 4;         // 0: lanes 0-15, 1: lanes 16-31
  const int l16     = lane & 15;

  v8f acc = {};
#pragma unroll
  for (int kk = 0; kk < FIN; kk += 32) {
    // A fragment: 16x32 bf16, contiguous K:
    // elems 0..15 -> K = kk + half*16 + i
    const float* arow = A + (long)(row0 + l16) * lda + kk + half * 16;
    const float4 a0 = *(const float4*)(arow + 0);
    const float4 a1 = *(const float4*)(arow + 4);
    const float4 a2 = *(const float4*)(arow + 8);
    const float4 a3 = *(const float4*)(arow + 12);
    v16bf a;
    a[0]  = (__bf16)a0.x; a[1]  = (__bf16)a0.y; a[2]  = (__bf16)a0.z; a[3]  = (__bf16)a0.w;
    a[4]  = (__bf16)a1.x; a[5]  = (__bf16)a1.y; a[6]  = (__bf16)a1.z; a[7]  = (__bf16)a1.w;
    a[8]  = (__bf16)a2.x; a[9]  = (__bf16)a2.y; a[10] = (__bf16)a2.z; a[11] = (__bf16)a2.w;
    a[12] = (__bf16)a3.x; a[13] = (__bf16)a3.y; a[14] = (__bf16)a3.z; a[15] = (__bf16)a3.w;

    // B fragment: 32x16 bf16, column = colTile*16 + l16, K = kk + half*16 + i
    const float* wcol = W + (long)(kk + half * 16) * ldw + colTile * 16 + l16;
    v16bf b;
#pragma unroll
    for (int i = 0; i < 16; ++i) b[i] = (__bf16)wcol[(long)i * ldw];

    acc = __builtin_amdgcn_wmma_f32_16x16x32_bf16(false, a, false, b,
                                                  (short)0, acc, false, false);
  }

  // C/D layout: lane<16 -> N=lane, M=vgpr ; lane>=16 -> N=lane-16, M=8+vgpr
  const int colN = colTile * 16 + l16;
  const float bv = bias ? bias[colN] : 0.0f;
#pragma unroll
  for (int i = 0; i < 8; ++i) {
    const int m = half * 8 + i;
    C[(long)(row0 + m) * ldc + coff + colN] = acc[i] + bv;
  }
}

__global__ void k_fill0(float* __restrict__ p, long n) {
  long i = (long)blockIdx.x * blockDim.x + threadIdx.x;
  long s = (long)gridDim.x * blockDim.x;
  for (; i < n; i += s) p[i] = 0.0f;
}

__global__ void k_degree(const int* __restrict__ dst, float* __restrict__ deg, int e) {
  int i = blockIdx.x * blockDim.x + threadIdx.x;
  if (i < e) atomicAdd(&deg[dst[i]], 1.0f);
}

__global__ void k_dinv(float* __restrict__ deg, int n) {
  int i = blockIdx.x * blockDim.x + threadIdx.x;
  if (i < n) {
    float d = deg[i];
    deg[i] = d > 0.0f ? 1.0f / sqrtf(d) : 0.0f;
  }
}

__global__ void k_norm(const int* __restrict__ src, const int* __restrict__ dst,
                       const float* __restrict__ dinv, float* __restrict__ nrm, int e) {
  int i = blockIdx.x * blockDim.x + threadIdx.x;
  if (i < e) nrm[i] = dinv[src[i]] * dinv[dst[i]];
}

// out[dst[e], :] += feat[src[e], :] * nrm[e] ; vec4 = (K*F)/4 float4 chunks per node
__global__ void k_prop(const float* __restrict__ feat,
                       const int* __restrict__ src, const int* __restrict__ dst,
                       const float* __restrict__ nrm,
                       float* __restrict__ out, int vec4, int e) {
  long i = (long)blockIdx.x * blockDim.x + threadIdx.x;
  long tot = (long)e * vec4;
  if (i >= tot) return;
  int ed = (int)(i / vec4);
  int c  = (int)(i % vec4);
  float w = nrm[ed];
  const float4 v = ((const float4*)(feat + (long)src[ed] * vec4 * 4))[c];
  float* o = out + (long)dst[ed] * (long)vec4 * 4 + (long)c * 4;
  atomicAdd(o + 0, v.x * w);
  atomicAdd(o + 1, v.y * w);
  atomicAdd(o + 2, v.z * w);
  atomicAdd(o + 3, v.w * w);
}

__global__ void k_combine(const float* __restrict__ a, const float* __restrict__ r,
                          float* __restrict__ o, long n) {
  long i = (long)blockIdx.x * blockDim.x + threadIdx.x;
  if (i < n) o[i] = fmaxf(a[i] + r[i], 0.0f);
}

// mean over K=2 stacks (optionally + outer relu)
__global__ void k_mean(const float* __restrict__ act, float* __restrict__ o,
                       int F, int do_relu, long nf) {
  long i = (long)blockIdx.x * blockDim.x + threadIdx.x;
  if (i >= nf) return;
  long node = i / F;
  int  f    = (int)(i % F);
  float v = 0.5f * (act[node * (long)(2 * F) + f] + act[node * (long)(2 * F) + F + f]);
  o[i] = do_relu ? fmaxf(v, 0.0f) : v;
}

extern "C" void kernel_launch(void* const* d_in, const int* in_sizes, int n_in,
                              void* d_out, int out_size, void* d_ws, size_t ws_size,
                              hipStream_t stream) {
  const float* x        = (const float*)d_in[0];
  const int*   ei       = (const int*)d_in[1];
  const int*   srcI     = ei;
  const int*   dstI     = ei + EE;
  const float* init_w1  = (const float*)d_in[2];
  const float* w1       = (const float*)d_in[3];
  const float* root_w1  = (const float*)d_in[4];
  const float* b1       = (const float*)d_in[5];
  const float* init_w2  = (const float*)d_in[6];
  const float* w2       = (const float*)d_in[7];
  const float* root_w2  = (const float*)d_in[8];
  const float* b2       = (const float*)d_in[9];
  float* outp = (float*)d_out;

  float* ws   = (float*)d_ws;
  float* deg  = ws;                               // N
  float* nrm  = deg + NN;                         // E
  float* root = nrm + EE;                         // N*K*64 (max)
  float* T0   = root + (long)NN * KSTACK * 64;    // N*K*64
  float* AG   = T0   + (long)NN * KSTACK * 64;    // N*K*64
  float* h1   = AG   + (long)NN * KSTACK * 64;    // N*64

  // ---- gcn_norm ----
  k_fill0 <<<512, 256, 0, stream>>>(deg, (long)NN);
  k_degree<<<(EE + 255) / 256, 256, 0, stream>>>(dstI, deg, EE);
  k_dinv  <<<(NN + 255) / 256, 256, 0, stream>>>(deg, NN);
  k_norm  <<<(EE + 255) / 256, 256, 0, stream>>>(srcI, dstI, deg, nrm, EE);

  // ---- one ARMA conv (shared weights, T=2) ----
  // FIN is a template parameter; runtime fin must match it.
  auto conv = [&](auto fin_tag, const float* X, int fout,
                  const float* initW, const float* wW, const float* rootW,
                  const float* bb, float* outbuf, int relu_out) {
    constexpr int FIN = decltype(fin_tag)::value;
    dim3 blk(32, fout / 16);
    const int  ldc  = KSTACK * fout;
    const long nkf  = (long)NN * KSTACK * fout;
    const int  vec4 = (KSTACK * fout) / 4;         // float4 chunks per node
    const long tot  = (long)EE * vec4;
    const unsigned pgrid = (unsigned)((tot + 255) / 256);
    const unsigned egrid = (unsigned)((nkf + 255) / 256);

    for (int k = 0; k < KSTACK; ++k) {
      k_gemm_wmma<FIN><<<NN / 16, blk, 0, stream>>>(
          X, FIN, rootW + (long)k * FIN * fout, fout, bb + k * fout,
          root, ldc, k * fout);
      k_gemm_wmma<FIN><<<NN / 16, blk, 0, stream>>>(
          X, FIN, initW + (long)k * FIN * fout, fout, nullptr,
          T0, ldc, k * fout);
    }
    // t = 0: propagate T0 -> AG, combine -> T0
    k_fill0  <<<2048, 256, 0, stream>>>(AG, nkf);
    k_prop   <<<pgrid, 256, 0, stream>>>(T0, srcI, dstI, nrm, AG, vec4, EE);
    k_combine<<<egrid, 256, 0, stream>>>(AG, root, T0, nkf);
    // t = 1: transform T0 -> AG (per-stack shared W, FIN == fout of this conv)
    if (fout == 64) {
      for (int k = 0; k < KSTACK; ++k)
        k_gemm_wmma<64><<<NN / 16, blk, 0, stream>>>(
            T0 + k * fout, ldc, wW + (long)k * fout * fout, fout, nullptr,
            AG, ldc, k * fout);
    } else {
      for (int k = 0; k < KSTACK; ++k)
        k_gemm_wmma<32><<<NN / 16, blk, 0, stream>>>(
            T0 + k * fout, ldc, wW + (long)k * fout * fout, fout, nullptr,
            AG, ldc, k * fout);
    }
    // propagate AG -> T0, combine -> AG
    k_fill0  <<<2048, 256, 0, stream>>>(T0, nkf);
    k_prop   <<<pgrid, 256, 0, stream>>>(AG, srcI, dstI, nrm, T0, vec4, EE);
    k_combine<<<egrid, 256, 0, stream>>>(T0, root, AG, nkf);
    // mean over stacks
    const long nf = (long)NN * fout;
    k_mean<<<(unsigned)((nf + 255) / 256), 256, 0, stream>>>(AG, outbuf, fout,
                                                             relu_out, nf);
  };

  conv(std::integral_constant<int, 64>{}, x,  64,
       init_w1, w1, root_w1, b1, h1,   1);   // relu(conv1)
  conv(std::integral_constant<int, 64>{}, h1, 32,
       init_w2, w2, root_w2, b2, outp, 0);   // conv2 -> d_out
}
